// AttentionAggregator_18588618457473
// MI455X (gfx1250) — matrix operand order repaired and checked by reading, not compiled, not run on hardware
//
#include <hip/hip_runtime.h>

typedef __attribute__((ext_vector_type(16))) __bf16 v16bf;
typedef __attribute__((ext_vector_type(4)))  __bf16 v4bf;
typedef __attribute__((ext_vector_type(8)))  float  v8f;

#define T_TOK    8192
#define D_DIM    768
#define W_MAX    32
#define KT_STEPS 24     // 768 / 32
#define NT_TILES 48     // 768 / 16
#define BLOCK    256    // 8 wave32

// ---------------------------------------------------------------------------
// Prep: swizzle Wq (f32, row-major [e][d]) into bf16 WMMA B-fragment layout.
// B[k][n] = Wq[n][k].  Fragment storage: [kt][nt][lane][e] contiguous, so the
// GEMM inner loop loads 32 contiguous bytes per lane per WMMA.
// Lane layout (16-bit B 32x16, this code's contiguous k order, same as A): n = lane&15, k = (lane>>4)*16 + e.
// ---------------------------------------------------------------------------
__global__ __launch_bounds__(BLOCK)
void prep_wq_kernel(const float* __restrict__ Wq, unsigned short* __restrict__ wqB_) {
  __bf16* wqB = (__bf16*)wqB_;
  int o = blockIdx.x * BLOCK + threadIdx.x;
  if (o >= D_DIM * D_DIM) return;
  int e  = o & 15;
  int l  = (o >> 4) & 31;
  int t  = o >> 9;
  int nt = t % NT_TILES;
  int kt = t / NT_TILES;
  int k  = kt * 32 + (l >> 4) * 16 + e;
  int nn = nt * 16 + (l & 15);
  wqB[o] = (__bf16)Wq[(size_t)nn * D_DIM + k];
}

// A-fragment (16-bit A 16x32, ISA 7.12.2): lane m = lane&15, half h = lane>>4.
// Elements 0..15 <-> k = h*16 + idx (contiguous 32B in LDS), matching B
__device__ __forceinline__ v16bf load_a_frag(const __bf16* tile, int lane) {
  const int m = lane & 15;
  const int h = lane >> 4;
  const __bf16* p = tile + m * D_DIM + h * 16;
  v16bf a;
#pragma unroll
  for (int e = 0; e < 16; ++e) a[e] = p[e];
  return a;
}

// B-fragment for B = S^T where S is row-major [w][d] in LDS:
// B[k][n] = S[n][k]; n = lane&15, k = k0 + (lane>>4)*16 + e  -> contiguous 32B.
__device__ __forceinline__ v16bf load_b_frag_trans(const __bf16* S, int n0, int k0, int lane) {
  const __bf16* p = S + (n0 + (lane & 15)) * D_DIM + k0 + ((lane >> 4) << 4);
  return *(const v16bf*)p;
}

// ---------------------------------------------------------------------------
// One workgroup per span. 256 threads = 8 wave32.
// LDS: sS 48K | sQ 48K | sc 4K | sc2 4K | sps/spo  => 106752 B (3 WG / WGP)
// ---------------------------------------------------------------------------
template <bool USE_PREP>
__global__ __launch_bounds__(BLOCK)
void span_attn_kernel(const float* __restrict__ emb,
                      const int*  __restrict__ spans,
                      const float* __restrict__ Wq,
                      const float* __restrict__ bq,
                      const unsigned short* __restrict__ wqB_,
                      float* __restrict__ out) {
  extern __shared__ char smem[];
  __bf16* sS  = (__bf16*)(smem);            // 32 x 768 bf16  (48 KB)
  __bf16* sQ  = (__bf16*)(smem + 49152);    // 32 x 768 bf16  (48 KB)
  float*  sc  = (float*) (smem + 98304);    // 32 x 32  f32   (4 KB)
  float*  sc2 = (float*) (smem + 102400);   // 4 x 16 x 16 f32 partials (4 KB)
  float*  sps = (float*) (smem + 106496);   // 32 f32
  int*    spo = (int*)   (smem + 106624);   // 32 int

  const __bf16* wqB = (const __bf16*)wqB_;
  const int n    = blockIdx.x;
  const int tid  = threadIdx.x;
  const int lane = tid & 31;
  const int wave = tid >> 5;

  const int s0  = spans[2 * n];
  const int len = spans[2 * n + 1] - s0 + 1;
  const float NEG_INF = -__builtin_inff();

  // ---- Phase 1: gather span embeddings -> LDS bf16, zero invalid rows ----
  if (tid < W_MAX) {
    int p = s0 + tid;
    spo[tid] = p < (T_TOK - 1) ? p : (T_TOK - 1);
  }
  for (int w = wave; w < W_MAX; w += 8) {
    int p = s0 + w;
    p = p < (T_TOK - 1) ? p : (T_TOK - 1);
    const float4* src = (const float4*)(emb + (size_t)p * D_DIM);
    v4bf* dst = (v4bf*)(sS + w * D_DIM);
    if (w < len) {
#pragma unroll 1
      for (int d4 = lane; d4 < D_DIM / 4; d4 += 32) {
        float4 f = src[d4];
        v4bf h;
        h[0] = (__bf16)f.x; h[1] = (__bf16)f.y; h[2] = (__bf16)f.z; h[3] = (__bf16)f.w;
        dst[d4] = h;
      }
    } else {
      v4bf z = {};
#pragma unroll 1
      for (int d4 = lane; d4 < D_DIM / 4; d4 += 32) dst[d4] = z;
    }
  }
  __syncthreads();

  // ---- Phase 2: Q = S @ Wq^T + bq  (bf16 WMMA, f32 accum) -> sQ ----
  // Each wave owns 6 N-tiles; each B fragment is loaded ONCE and used for
  // both M-tiles (2 accumulators) -> halves Wq L2 traffic.
#pragma unroll 1
  for (int i = 0; i < 6; ++i) {
    const int nt = wave * 6 + i;          // 0..47
    v8f acc0 = {};
    v8f acc1 = {};
    const __bf16* bptr = wqB + ((size_t)(nt * 32 + lane) << 4);
#pragma unroll 1
    for (int kt = 0; kt < KT_STEPS; ++kt) {
      v16bf b;
      if (USE_PREP) {
        b = *(const v16bf*)bptr;
        bptr += (size_t)NT_TILES * 32 * 16;     // next kt slab
      } else {
        const float* p = Wq + (size_t)(nt * 16 + (lane & 15)) * D_DIM + kt * 32 + ((lane >> 4) << 4);
#pragma unroll
        for (int e = 0; e < 16; ++e) b[e] = (__bf16)p[e];
      }
      v16bf a0 = load_a_frag(sS + kt * 32, lane);
      v16bf a1 = load_a_frag(sS + 16 * D_DIM + kt * 32, lane);
      acc0 = __builtin_amdgcn_wmma_f32_16x16x32_bf16(false, a0, false, b, (short)0, acc0,
                                                     false, false);
      acc1 = __builtin_amdgcn_wmma_f32_16x16x32_bf16(false, a1, false, b, (short)0, acc1,
                                                     false, false);
    }
    const int col   = nt * 16 + (lane & 15);
    const int rbase = (lane >> 4) << 3;     // C/D: VGPR r -> row r (+8 for hi lanes)
    const float bqv = bq[col];
#pragma unroll
    for (int r = 0; r < 8; ++r)
      sQ[(rbase + r) * D_DIM + col] = (__bf16)(acc0[r] + bqv);
#pragma unroll
    for (int r = 0; r < 8; ++r)
      sQ[(16 + rbase + r) * D_DIM + col] = (__bf16)(acc1[r] + bqv);
  }
  __syncthreads();

  // ---- Phase 3: scores = Q @ S^T (32x32, K=768), K split over 8 waves ----
  {
    const int tile = wave & 3;          // 4 output tiles: mt = tile>>1, nt = tile&1
    const int kh   = wave >> 2;         // K half: kt in [kh*12, kh*12+12)
    const int mt   = tile >> 1;
    const int nt   = tile & 1;
    v8f acc = {};
#pragma unroll 1
    for (int kt = kh * 12; kt < kh * 12 + 12; ++kt) {
      v16bf a = load_a_frag(sQ + mt * 16 * D_DIM + kt * 32, lane);
      v16bf b = load_b_frag_trans(sS, nt * 16, kt * 32, lane);
      acc = __builtin_amdgcn_wmma_f32_16x16x32_bf16(false, a, false, b, (short)0, acc,
                                                    false, false);
    }
    const int rloc = (lane >> 4) << 3;
    const int cloc = lane & 15;
    if (kh == 1) {
#pragma unroll
      for (int r = 0; r < 8; ++r)
        sc2[tile * 256 + (rloc + r) * 16 + cloc] = acc[r];
    }
    __syncthreads();
    if (kh == 0) {
      const int col   = nt * 16 + cloc;
      const int rbase = mt * 16 + rloc;
#pragma unroll
      for (int r = 0; r < 8; ++r)
        sc[(rbase + r) * W_MAX + col] = acc[r] + sc2[tile * 256 + (rloc + r) * 16 + cloc];
    }
  }
  __syncthreads();

  // ---- Phase 4: masked softmax per row q, then sum over valid q rows ----
  if (tid < W_MAX) {
    const int q = tid;
    float row[W_MAX];
    float m = NEG_INF;
#pragma unroll
    for (int k = 0; k < W_MAX; ++k) {
      float v = (k < len) ? sc[q * W_MAX + k] : NEG_INF;
      row[k] = v;
      m = fmaxf(m, v);
    }
    float s = 0.0f;
#pragma unroll
    for (int k = 0; k < W_MAX; ++k) {
      float v = (k < len) ? __expf(row[k] - m) : 0.0f;
      row[k] = v;
      s += v;
    }
    const float inv = 1.0f / s;
#pragma unroll
    for (int k = 0; k < W_MAX; ++k) sc[q * W_MAX + k] = row[k] * inv;
  }
  __syncthreads();
  if (tid < W_MAX) {
    const int k = tid;
    float ps = 0.0f;
#pragma unroll
    for (int q = 0; q < W_MAX; ++q)
      if (q < len) ps += sc[q * W_MAX + k];
    sps[k] = ps;   // == 0 for invalid k (softmax weight 0)
  }
  __syncthreads();

  // ---- Phase 5: out[d] = sum_k psum[k] * emb_f32[pos[k]][d] (full f32) ----
  if (tid < D_DIM / 4) {
    float4 acc = make_float4(0.f, 0.f, 0.f, 0.f);
#pragma unroll 1
    for (int k = 0; k < W_MAX; ++k) {
      const float w = sps[k];
      const float4 f = *(const float4*)(emb + (size_t)spo[k] * D_DIM + tid * 4);
      acc.x = fmaf(w, f.x, acc.x);
      acc.y = fmaf(w, f.y, acc.y);
      acc.z = fmaf(w, f.z, acc.z);
      acc.w = fmaf(w, f.w, acc.w);
    }
    *(float4*)(out + (size_t)n * D_DIM + tid * 4) = acc;
  }
}

extern "C" void kernel_launch(void* const* d_in, const int* in_sizes, int n_in,
                              void* d_out, int out_size, void* d_ws, size_t ws_size,
                              hipStream_t stream) {
  (void)n_in; (void)out_size;
  const float* emb   = (const float*)d_in[0];   // (8192, 768) f32
  const int*   spans = (const int*)d_in[1];     // (N, 2) i32
  const float* Wq    = (const float*)d_in[2];   // (768, 768) f32
  const float* bq    = (const float*)d_in[3];   // (768,) f32
  float* out = (float*)d_out;                   // (N, 768) f32

  const int nspan = in_sizes[1] / 2;
  const size_t need = (size_t)D_DIM * D_DIM * 2;       // bf16 swizzled Wq
  const bool use_prep = (d_ws != nullptr) && (ws_size >= need);
  const size_t smem_bytes = 49152 + 49152 + 4096 + 4096 + 128 + 128;  // 106752 B

  if (use_prep) {
    unsigned short* wqB = (unsigned short*)d_ws;
    prep_wq_kernel<<<(D_DIM * D_DIM + BLOCK - 1) / BLOCK, BLOCK, 0, stream>>>(Wq, wqB);
    span_attn_kernel<true><<<nspan, BLOCK, smem_bytes, stream>>>(emb, spans, Wq, bq, wqB, out);
  } else {
    span_attn_kernel<false><<<nspan, BLOCK, smem_bytes, stream>>>(emb, spans, Wq, bq, nullptr, out);
  }
}
